// MultipleHeadSelfAttention_17600775979546
// MI455X (gfx1250) — matrix operand order repaired and checked by reading, not compiled, not run on hardware
//
#include <hip/hip_runtime.h>
#include <hip/hip_bf16.h>
#include <math.h>

typedef _Float16 half_t;
typedef _Float16 v16h __attribute__((ext_vector_type(16)));
typedef _Float16 v8h  __attribute__((ext_vector_type(8)));
typedef float    v8f  __attribute__((ext_vector_type(8)));

#define D_MODEL   1024
#define NUM_HEADS 16
#define HEAD_DIM  64
#define BATCH     2
#define SEQ       2048
#define ROWS      (BATCH * SEQ)     /* 4096 */
#define KQV_COLS  (3 * D_MODEL)     /* 3072 */

union V16 { v16h v; v8h h[2]; };

// ---- A-fragment (16x32 f16, MxK): row_ptr = this lane's row (m = lane&15), K-contiguous.
// Slots 0-7 -> K = (lane>=16?8:0)+j ; slots 8-15 -> K = 16 + (lane>=16?8:0)+j.
__device__ inline v16h load_a_frag(const half_t* row_ptr, int lane) {
  const int hi = (lane >> 4) << 3;
  V16 r;
  r.h[0] = *(const v8h*)(row_ptr + hi);
  r.h[1] = *(const v8h*)(row_ptr + 16 + hi);
  return r.v;
}

// ---- B-fragment (32x16 f16, KxN): col_ptr = this lane's column (n = lane&15), K-contiguous.
// Slots 0-7 -> K = (lane>=16?8:0)+j ; slots 8-15 -> K = 16 + (lane>=16?8:0)+j.
__device__ inline v16h load_b_frag(const half_t* col_ptr, int lane) {
  const half_t* p = col_ptr + ((lane >> 4) << 3);
  V16 r;
  r.h[0] = *(const v8h*)(p);
  r.h[1] = *(const v8h*)(p + 16);
  return r.v;
}

__device__ inline v8f wmma_f16(v16h a, v16h b, v8f c) {
  return __builtin_amdgcn_wmma_f32_16x16x32_f16(false, a, false, b, (short)0, c, false, false);
}

// ---- gfx1250 async global->LDS copy (16B per lane), tracked by ASYNCcnt.
__device__ inline void async_copy_b128(void* lds_ptr, const void* gptr) {
  const unsigned lds_off = (unsigned)(unsigned long long)(uintptr_t)lds_ptr; // LDS aperture: offset in [31:0]
  asm volatile("global_load_async_to_lds_b128 %0, %1, off"
               :: "v"(lds_off), "v"((unsigned long long)(uintptr_t)gptr)
               : "memory");
}
__device__ inline void wait_asynccnt0() {
  asm volatile("s_wait_asynccnt 0x0" ::: "memory");
}

// ------------------------------------------------------------------ convert
__global__ void f32_to_f16_kernel(const float* __restrict__ src,
                                  half_t* __restrict__ dst, int n) {
  int i = blockIdx.x * blockDim.x + threadIdx.x;
  int stride = gridDim.x * blockDim.x;
  for (; i < n; i += stride) dst[i] = (half_t)src[i];
}

// ---------------------------------------------------- fused QKV GEMM + RoPE
// C = x_f16 (4096x1024) @ Wkqv_f16^T (1024x3072). Each wave: 32 rows x 64 cols
// (2 A-fragments reuse 4 B-fragments -> 8 WMMA per K-step).
__global__ __launch_bounds__(256) void qkv_rope_kernel(
    const half_t* __restrict__ xh, const half_t* __restrict__ wh,
    half_t* __restrict__ Qh, half_t* __restrict__ Kh, half_t* __restrict__ Vth) {
  const int lane = threadIdx.x & 31;
  const int wave = threadIdx.x >> 5;
  const int r0 = (blockIdx.y * 8 + wave) * 32;   // gridDim.y = ROWS/256 = 16
  const int c0 = blockIdx.x * 64;                // gridDim.x = KQV_COLS/64 = 48
  const int n  = lane & 15;

  v8f acc[2][4];
#pragma unroll
  for (int rr = 0; rr < 2; ++rr)
#pragma unroll
    for (int t = 0; t < 4; ++t) acc[rr][t] = v8f{};

  const half_t* arow0 = xh + (size_t)(r0 + n) * D_MODEL;
  const half_t* arow1 = xh + (size_t)(r0 + 16 + n) * D_MODEL;
  const half_t* brow[4];
#pragma unroll
  for (int t = 0; t < 4; ++t)
    brow[t] = wh + (size_t)(c0 + t * 16 + n) * D_MODEL;

  for (int k = 0; k < D_MODEL; k += 32) {
    v16h a0 = load_a_frag(arow0 + k, lane);
    v16h a1 = load_a_frag(arow1 + k, lane);
#pragma unroll
    for (int t = 0; t < 4; ++t) {
      v16h bf = load_b_frag(brow[t] + k, lane);
      acc[0][t] = wmma_f16(a0, bf, acc[0][t]);
      acc[1][t] = wmma_f16(a1, bf, acc[1][t]);
    }
  }

  const int sec = c0 >> 10;                 // 0=K, 1=Q, 2=V (jnp.split order)
  const int h   = (c0 & 1023) >> 6;
  const int rowbase = (lane >> 4) << 3;

#pragma unroll
  for (int rr = 0; rr < 2; ++rr) {
    const int rbase = r0 + rr * 16 + rowbase;
    if (sec == 2) {
      // V: store f16 transposed  Vt[(b*H+h)*64 + dd][pos]
#pragma unroll
      for (int t = 0; t < 4; ++t) {
        const int dd = t * 16 + n;
#pragma unroll
        for (int v = 0; v < 8; ++v) {
          const int r = rbase + v;
          const int b = r >> 11, pos = r & (SEQ - 1);
          Vth[((size_t)(b * NUM_HEADS + h) * HEAD_DIM + dd) * SEQ + pos] =
              (half_t)acc[rr][t][v];
        }
      }
    } else {
      half_t* dst = (sec == 0) ? Kh : Qh;
      const float ln_theta_over_half = 0.28782313662425575f;  // ln(10000)/32
#pragma unroll
      for (int t = 0; t < 4; ++t) {
        const int dd = t * 16 + n;
        const float invf = __expf(-(float)(dd >> 1) * ln_theta_over_half);
        const bool even = (dd & 1) == 0;
#pragma unroll
        for (int v = 0; v < 8; ++v) {
          const int r = rbase + v;
          const int b = r >> 11, pos = r & (SEQ - 1);
          float self = acc[rr][t][v];
          float partner = __shfl_xor(self, 1, 32);  // pair element (dd ^ 1)
          float s, c;
          __sincosf((float)pos * invf, &s, &c);
          float out = even ? (self * c - partner * s)   // r1 = x1*cos - x2*sin
                           : (partner * s + self * c);  // r2 = x1*sin + x2*cos
          dst[((size_t)(b * NUM_HEADS + h) * SEQ + pos) * HEAD_DIM + dd] =
              (half_t)out;
        }
      }
    }
  }
}

// --------------------------------------------------------- flash attention
// 8 waves / workgroup, 128 query rows. 32-key blocks of K and V^T are staged
// into LDS once per workgroup with double-buffered async global->LDS copies.
__global__ __launch_bounds__(256) void attn_kernel(
    const half_t* __restrict__ Qh, const half_t* __restrict__ Kh,
    const half_t* __restrict__ Vth, half_t* __restrict__ Oh) {
  __shared__ __align__(16) half_t Kbuf[2][32 * HEAD_DIM];  // [key_local][d]
  __shared__ __align__(16) half_t Vbuf[2][HEAD_DIM * 32];  // [dd][key_local]
  __shared__ __align__(16) half_t Plds[8][16 * 32];        // per-wave P staging

  const int tid  = threadIdx.x;
  const int lane = tid & 31;
  const int wave = tid >> 5;
  const int qb0 = blockIdx.x * 128;
  const int q0  = qb0 + wave * 16;
  const int h   = blockIdx.y;
  const int b   = blockIdx.z;
  const size_t bh = (size_t)(b * NUM_HEADS + h);
  const half_t* Qb = Qh + bh * SEQ * HEAD_DIM;
  const half_t* Kb = Kh + bh * SEQ * HEAD_DIM;
  const half_t* Vt = Vth + bh * HEAD_DIM * SEQ;
  const int n = lane & 15;
  const int rowbase = (lane >> 4) << 3;

  // cooperative loader mapping: 256 threads x 16B = 4KB per issue
  const int kl = tid >> 3, dc = (tid & 7) << 3;   // K tile: 32 keys x 64 d
  const int dv = tid >> 2, kc = (tid & 3) << 3;   // V tile: 64 dd  x 32 keys

  v16h qa0 = load_a_frag(Qb + (size_t)(q0 + n) * HEAD_DIM, lane);
  v16h qa1 = load_a_frag(Qb + (size_t)(q0 + n) * HEAD_DIM + 32, lane);

  v8f acc[4] = {v8f{}, v8f{}, v8f{}, v8f{}};
  float m_i[8], l_i[8];
#pragma unroll
  for (int v = 0; v < 8; ++v) { m_i[v] = -1e30f; l_i[v] = 0.f; }
  const float scale = 0.125f;  // 1/sqrt(64)
  half_t* myP = &Plds[wave][0];

  const int nkb = blockIdx.x * 4 + 4;  // (qb0+128)/32 key blocks for this workgroup

  // prologue: request block 0 into buffer 0
  async_copy_b128(&Kbuf[0][kl * HEAD_DIM + dc], Kb + (size_t)kl * HEAD_DIM + dc);
  async_copy_b128(&Vbuf[0][dv * 32 + kc],       Vt + (size_t)dv * SEQ + kc);

  for (int ib = 0; ib < nkb; ++ib) {
    const int kb  = ib * 32;
    const int buf = ib & 1;
    wait_asynccnt0();     // my copies for block ib have landed in LDS
    __syncthreads();      // everyone's copies landed; everyone done reading buf^1
    if (ib + 1 < nkb) {   // overlap next block's copies with this block's math
      const int kb2 = kb + 32, buf2 = buf ^ 1;
      async_copy_b128(&Kbuf[buf2][kl * HEAD_DIM + dc],
                      Kb + (size_t)(kb2 + kl) * HEAD_DIM + dc);
      async_copy_b128(&Vbuf[buf2][dv * 32 + kc],
                      Vt + (size_t)dv * SEQ + kb2 + kc);
    }
    if (kb < q0 + 16) {   // wave-uniform causal skip
      const half_t* Kl = Kbuf[buf];
      const half_t* Vl = Vbuf[buf];
      v8f S[2];
#pragma unroll
      for (int sub = 0; sub < 2; ++sub) {
        const int keyl = sub * 16 + n;
        const int key  = kb + keyl;
        v8f s = {};
        s = wmma_f16(qa0, load_b_frag(Kl + keyl * HEAD_DIM, lane), s);
        s = wmma_f16(qa1, load_b_frag(Kl + keyl * HEAD_DIM + 32, lane), s);
#pragma unroll
        for (int v = 0; v < 8; ++v) {
          const int q = q0 + rowbase + v;
          s[v] = (key <= q) ? s[v] * scale : -1e30f;  // causal mask
        }
        S[sub] = s;
      }
      // online softmax: row reductions across 16 lanes of each half-wave
      float alpha[8];
#pragma unroll
      for (int v = 0; v < 8; ++v) {
        float mx = fmaxf(S[0][v], S[1][v]);
#pragma unroll
        for (int off = 8; off >= 1; off >>= 1) mx = fmaxf(mx, __shfl_xor(mx, off, 32));
        const float m_new = fmaxf(m_i[v], mx);
        const float a  = __expf(m_i[v] - m_new);
        const float p0 = __expf(S[0][v] - m_new);
        const float p1 = __expf(S[1][v] - m_new);
        S[0][v] = p0; S[1][v] = p1;
        float rs = p0 + p1;
#pragma unroll
        for (int off = 8; off >= 1; off >>= 1) rs += __shfl_xor(rs, off, 32);
        l_i[v] = l_i[v] * a + rs;
        m_i[v] = m_new;
        alpha[v] = a;
      }
#pragma unroll
      for (int t = 0; t < 4; ++t)
#pragma unroll
        for (int v = 0; v < 8; ++v) acc[t][v] *= alpha[v];

      // stage P (accumulator layout) -> per-wave LDS 16x32 -> A-fragment
      // (same-wave LDS ops are in-order; no cross-wave readers)
#pragma unroll
      for (int v = 0; v < 8; ++v) {
        myP[(rowbase + v) * 32 + n]      = (half_t)S[0][v];
        myP[(rowbase + v) * 32 + 16 + n] = (half_t)S[1][v];
      }
      v16h pa = load_a_frag(myP + n * 32, lane);
#pragma unroll
      for (int t = 0; t < 4; ++t) {
        v16h vb = load_b_frag(Vl + (t * 16 + n) * 32, lane);
        acc[t] = wmma_f16(pa, vb, acc[t]);
      }
    }
  }
  // normalize and store attention output f16 as (B*L, 1024) for the projection
#pragma unroll
  for (int v = 0; v < 8; ++v) {
    const float inv = 1.0f / l_i[v];
    const int q = q0 + rowbase + v;
    const size_t base = ((size_t)b * SEQ + q) * D_MODEL + h * HEAD_DIM;
#pragma unroll
    for (int t = 0; t < 4; ++t)
      Oh[base + t * 16 + n] = (half_t)(acc[t][v] * inv);
  }
}

// -------------------------------------------------------- output projection
__global__ __launch_bounds__(256) void proj_kernel(
    const half_t* __restrict__ Oh, const half_t* __restrict__ Woh,
    float* __restrict__ out) {
  const int lane = threadIdx.x & 31;
  const int wave = threadIdx.x >> 5;
  const int r0 = (blockIdx.y * 8 + wave) * 32;  // gridDim.y = 16
  const int c0 = blockIdx.x * 64;               // gridDim.x = 16
  const int n  = lane & 15;

  v8f acc[2][4];
#pragma unroll
  for (int rr = 0; rr < 2; ++rr)
#pragma unroll
    for (int t = 0; t < 4; ++t) acc[rr][t] = v8f{};

  const half_t* arow0 = Oh + (size_t)(r0 + n) * D_MODEL;
  const half_t* arow1 = Oh + (size_t)(r0 + 16 + n) * D_MODEL;
  const half_t* brow[4];
#pragma unroll
  for (int t = 0; t < 4; ++t)
    brow[t] = Woh + (size_t)(c0 + t * 16 + n) * D_MODEL;

  for (int k = 0; k < D_MODEL; k += 32) {
    v16h a0 = load_a_frag(arow0 + k, lane);
    v16h a1 = load_a_frag(arow1 + k, lane);
#pragma unroll
    for (int t = 0; t < 4; ++t) {
      v16h bf = load_b_frag(brow[t] + k, lane);
      acc[0][t] = wmma_f16(a0, bf, acc[0][t]);
      acc[1][t] = wmma_f16(a1, bf, acc[1][t]);
    }
  }
  const int rowbase = (lane >> 4) << 3;
#pragma unroll
  for (int rr = 0; rr < 2; ++rr)
#pragma unroll
    for (int t = 0; t < 4; ++t)
#pragma unroll
      for (int v = 0; v < 8; ++v)
        out[(size_t)(r0 + rr * 16 + rowbase + v) * D_MODEL + c0 + t * 16 + n] =
            acc[rr][t][v];
}

// --------------------------------------------------------------------------
extern "C" void kernel_launch(void* const* d_in, const int* in_sizes, int n_in,
                              void* d_out, int out_size, void* d_ws, size_t ws_size,
                              hipStream_t stream) {
  (void)n_in; (void)out_size; (void)ws_size;
  const float* x    = (const float*)d_in[0];
  const float* Wkqv = (const float*)d_in[1];
  const float* Wo   = (const float*)d_in[2];

  char* ws = (char*)d_ws;
  size_t off = 0;
  auto carve = [&](size_t bytes) {
    char* p = ws + off;
    off += (bytes + 255) & ~(size_t)255;
    return p;
  };
  half_t* xh    = (half_t*)carve((size_t)ROWS * D_MODEL * sizeof(half_t));
  half_t* wkqvh = (half_t*)carve((size_t)KQV_COLS * D_MODEL * sizeof(half_t));
  half_t* woh   = (half_t*)carve((size_t)D_MODEL * D_MODEL * sizeof(half_t));
  half_t* Qh    = (half_t*)carve((size_t)BATCH * NUM_HEADS * SEQ * HEAD_DIM * sizeof(half_t));
  half_t* Kh    = (half_t*)carve((size_t)BATCH * NUM_HEADS * SEQ * HEAD_DIM * sizeof(half_t));
  half_t* Vth   = (half_t*)carve((size_t)BATCH * NUM_HEADS * HEAD_DIM * SEQ * sizeof(half_t));
  half_t* Oh    = (half_t*)carve((size_t)ROWS * D_MODEL * sizeof(half_t));

  const int nx = in_sizes[0], nw = in_sizes[1], no = in_sizes[2];
  f32_to_f16_kernel<<<dim3((nx + 255) / 256), 256, 0, stream>>>(x, xh, nx);
  f32_to_f16_kernel<<<dim3((nw + 255) / 256), 256, 0, stream>>>(Wkqv, wkqvh, nw);
  f32_to_f16_kernel<<<dim3((no + 255) / 256), 256, 0, stream>>>(Wo, woh, no);

  qkv_rope_kernel<<<dim3(KQV_COLS / 64, ROWS / 256), 256, 0, stream>>>(
      xh, wkqvh, Qh, Kh, Vth);

  attn_kernel<<<dim3(SEQ / 128, NUM_HEADS, BATCH), 256, 0, stream>>>(
      Qh, Kh, Vth, Oh);

  proj_kernel<<<dim3(D_MODEL / 64, ROWS / 256), 256, 0, stream>>>(
      Oh, woh, (float*)d_out);
}
